// INECN_Layer_Context_81088982548662
// MI455X (gfx1250) — matrix operand order repaired and checked by reading, not compiled, not run on hardware
//
#include <hip/hip_runtime.h>
#include <hip/hip_bf16.h>

// ---------------------------------------------------------------------------
// INECN interaction-network forward for MI455X (gfx1250, wave32, WMMA).
//
// One-hot R/S/H -> index extraction; per-edge MLPs as 16-edge WMMA tiles
// (v_wmma_f32_16x16x32_f16, f32 accumulate). Weights are staged into LDS
// PRE-SWIZZLED into the per-lane B-fragment layout so each fragment is one
// contiguous 32B run per lane (2x ds_load_b128). A fragments load as two
// aligned v8h chunks. Scatter via global_atomic_add_f32; node MLP scalar.
// ---------------------------------------------------------------------------

#define NNODES 300
#define NREL   89700                      // NNODES*NNODES - NNODES
#define NTILES ((NREL + 15) / 16)         // 5607 16-edge tiles
#define WPB    4                          // waves per block (edge kernels)
#define TPB    (WPB * 32)

typedef __attribute__((ext_vector_type(16))) _Float16 v16h;
typedef __attribute__((ext_vector_type(8)))  _Float16 v8h;
typedef __attribute__((ext_vector_type(8)))  float    v8f;

// ---- A fragment: 16x32 f16 (ISA 7.12.2). Per lane: two contiguous 8-half
// runs: row[hf*8 + 0..7] and row[hf*8 + 16..23]. Two 16B LDS loads. ---------
__device__ __forceinline__ v16h ldsAv(const _Float16* act, int stride, int ks, int lane) {
  const int hf = (lane >> 4) & 1;
  const int m  = lane & 15;
  const _Float16* row = act + m * stride + ks * 32 + hf * 8;
  v8h lo = *(const v8h*)(row);
  v8h hi = *(const v8h*)(row + 16);
  return __builtin_shufflevector(lo, hi, 0, 1, 2, 3, 4, 5, 6, 7,
                                 8, 9, 10, 11, 12, 13, 14, 15);
}

// ---- B fragment from pre-swizzled LDS: frag = ks*ntiles + nt, 32B per lane.
__device__ __forceinline__ v16h ldsBswz(const _Float16* wswz, int ntiles, int ks,
                                        int nt, int lane) {
  return *(const v16h*)(wswz + (size_t)(((ks * ntiles + nt) * 32 + lane) * 16));
}

// C/D 16x16 f32: VGPR r -> M = r + 8*hf, N = lane&15 (+16*nt).

// ---- wave tile GEMM: out[16][NPAD] = act[16][KPAD] @ W^T, bias(+relu) -----
template <int KPAD, int NPAD, bool RELU>
__device__ __forceinline__ void waveGemmStore(const _Float16* act, int actStride,
                                              const _Float16* wswz, const float* bias,
                                              _Float16* out, int outStride, int lane) {
  const int hf = (lane >> 4) & 1;
  const int nc = lane & 15;
#pragma unroll
  for (int nt = 0; nt < NPAD / 16; ++nt) {
    v8f c = {};
#pragma unroll
    for (int ks = 0; ks < KPAD / 32; ++ks) {
      v16h a = ldsAv(act, actStride, ks, lane);
      v16h b = ldsBswz(wswz, NPAD / 16, ks, nt, lane);
      c = __builtin_amdgcn_wmma_f32_16x16x32_f16(false, a, false, b, (short)0, c, false, false);
    }
    const int   n  = nt * 16 + nc;
    const float bv = bias[n];
#pragma unroll
    for (int r = 0; r < 8; ++r) {
      float v = c[r] + bv;
      if (RELU) v = fmaxf(v, 0.f);
      out[(r + 8 * hf) * outStride + n] = (_Float16)v;
    }
  }
}

// ---- weight staging: f32 -> f16, zero padded, PRE-SWIZZLED B layout -------
// dst[((ks*ntiles+nt)*32 + lane)*16 + i] = W[n = (lane&15)+nt*16][k = ks*32 + (lane>>4)*8 + (i&7) + (i>>3)*16]
__device__ void stageWswz(_Float16* dst, const float* src, int dout, int din,
                          int kpad, int npad, int tid) {
  const int ntiles = npad / 16;
  for (int idx = tid; idx < kpad * npad; idx += TPB) {
    const int frag = idx >> 9, rem = idx & 511;
    const int lane = rem >> 4, i = rem & 15;
    const int ks = frag / ntiles, nt = frag - ks * ntiles;
    const int k = ks * 32 + ((lane >> 4) & 1) * 8 + (i & 7) + (i >> 3) * 16;
    const int n = (lane & 15) + nt * 16;
    const float v = (k < din && n < dout) ? src[n * din + k] : 0.f;
    dst[idx] = (_Float16)v;
  }
}
__device__ void stageB(float* dst, const float* src, int dout, int npad, int tid) {
  for (int i = tid; i < npad; i += TPB) dst[i] = (i < dout) ? src[i] : 0.f;
}
// W1 = [w1_1(100x16) | w1_1(100x16) | w1_2(100x8)] -> kpad=64, npad=112, swizzled
__device__ void stageWcombSwz(_Float16* dst, const float* wa, const float* wb, int tid) {
  const int ntiles = 7;
  for (int idx = tid; idx < 64 * 112; idx += TPB) {
    const int frag = idx >> 9, rem = idx & 511;
    const int lane = rem >> 4, i = rem & 15;
    const int ks = frag / ntiles, nt = frag - ks * ntiles;
    const int k = ks * 32 + ((lane >> 4) & 1) * 8 + (i & 7) + (i >> 3) * 16;
    const int n = (lane & 15) + nt * 16;
    float v = 0.f;
    if (n < 100) {
      if (k < 16)      v = wa[n * 16 + k];
      else if (k < 32) v = wa[n * 16 + (k - 16)];
      else if (k < 40) v = wb[n * 8 + (k - 32)];
    }
    dst[idx] = (_Float16)v;
  }
}

// ---------------------------------------------------------------------------
// Kernel 1: extract sender/receiver/hyper indices from one-hot matrices.
// R is [NR,N] row-major; S,H are [N,NR].
// ---------------------------------------------------------------------------
__global__ void extract_idx_kernel(const float* __restrict__ R,
                                   const float* __restrict__ S,
                                   const float* __restrict__ H,
                                   int* __restrict__ recv, int* __restrict__ send,
                                   int* __restrict__ hidx) {
  const int total = NREL * NNODES;  // 26,910,000 < 2^31
  for (int i = blockIdx.x * blockDim.x + threadIdx.x; i < total;
       i += gridDim.x * blockDim.x) {
    if (R[i] != 0.f) recv[i / NNODES] = i % NNODES;
    if (S[i] != 0.f) send[i % NREL] = i / NREL;
    if (H[i] != 0.f) hidx[i % NREL] = i / NREL;
  }
}

// ---------------------------------------------------------------------------
// Kernel 2 (per phase): edge MLPs.
//  branch A (phi_E_O): x[40] ->50 relu ->50 relu ->32  => E_O, scatter+loss
//  branch B (W1comb):  x[40] ->100 relu ->(mlp_ee)32   => E_R, scatter
// ---------------------------------------------------------------------------
__global__ __launch_bounds__(TPB) void edge_kernel_A(
    const int* __restrict__ recv, const int* __restrict__ send,
    const int* __restrict__ hidx,
    const float* __restrict__ nodefeat,   // [NNODES][16]
    const float* __restrict__ edgefeat,   // [NREL][8]
    const float* A1w, const float* A1b,   // 50x40
    const float* A2w, const float* A2b,   // 50x50
    const float* A3w, const float* A3b,   // 32x50
    const float* w1a, const float* w1b, const float* b1v,  // 100x16,100x8,100
    const float* eew, const float* eeb,   // 32x100
    float* __restrict__ nodeAcc1, float* __restrict__ nodeAcc2,
    float* __restrict__ lossAcc) {
  __shared__ __align__(32) _Float16 sA1[64 * 64], sA2[64 * 64], sA3[64 * 32];
  __shared__ __align__(32) _Float16 sW1[64 * 112], sEE[128 * 32];
  __shared__ float sA1b[64], sA2b[64], sA3b[32], sB1[112], sEEb[32];
  __shared__ __align__(32) _Float16 sXb[WPB][16 * 64];
  __shared__ __align__(32) _Float16 sHb[WPB][16 * 128];

  const int tid = threadIdx.x;
  stageWswz(sA1, A1w, 50, 40, 64, 64, tid);
  stageWswz(sA2, A2w, 50, 50, 64, 64, tid);
  stageWswz(sA3, A3w, 32, 50, 64, 32, tid);
  stageWcombSwz(sW1, w1a, w1b, tid);
  stageWswz(sEE, eew, 32, 100, 128, 32, tid);
  stageB(sA1b, A1b, 50, 64, tid);
  stageB(sA2b, A2b, 50, 64, tid);
  stageB(sA3b, A3b, 32, 32, tid);
  stageB(sB1, b1v, 100, 112, tid);
  stageB(sEEb, eeb, 32, 32, tid);
  __syncthreads();

  const int lane = tid & 31;
  const int wv   = tid >> 5;
  const int gw   = blockIdx.x * WPB + wv;
  const int nw   = gridDim.x * WPB;
  const int hf   = (lane >> 4) & 1;
  const int nc   = lane & 15;
  _Float16* bufX = sXb[wv];
  _Float16* bufH = sHb[wv];
  unsigned int* bufX32 = (unsigned int*)bufX;
  unsigned int* bufH32 = (unsigned int*)bufH;
  float lossLocal = 0.f;

  for (int tile = gw; tile < NTILES; tile += nw) {
    const int e0 = tile * 16;
    // stage x = [nf[recv] | nf[send] | edgefeat], packed f16 pair stores
    for (int idx = lane; idx < 16 * 32; idx += 32) {
      const int row = idx >> 5, c0 = (idx & 31) << 1;
      const int e = e0 + row;
      float v0 = 0.f, v1 = 0.f;
      if (e < NREL) {
        if (c0 < 16)      { const float* p = &nodefeat[recv[e] * 16 + c0];
                            v0 = p[0]; v1 = p[1]; }
        else if (c0 < 32) { const float* p = &nodefeat[send[e] * 16 + (c0 - 16)];
                            v0 = p[0]; v1 = p[1]; }
        else if (c0 < 40) { const float* p = &edgefeat[e * 8 + (c0 - 32)];
                            v0 = p[0]; v1 = p[1]; }
      }
      union { _Float16 h[2]; unsigned int u; } pk;
      pk.h[0] = (_Float16)v0; pk.h[1] = (_Float16)v1;
      bufX32[idx] = pk.u;
    }
    // ---- branch A: MLP3 ----
    waveGemmStore<64, 64, true>(bufX, 64, sA1, sA1b, bufH, 128, lane);        // h1: cols 0-63
    waveGemmStore<64, 64, true>(bufH, 128, sA2, sA2b, bufH + 64, 128, lane);  // h2: cols 64-127
#pragma unroll
    for (int nt = 0; nt < 2; ++nt) {  // E_O = h2 @ A3^T + b  (N=32)
      v8f c = {};
#pragma unroll
      for (int ks = 0; ks < 2; ++ks) {
        v16h a = ldsAv(bufH + 64, 128, ks, lane);
        v16h b = ldsBswz(sA3, 2, ks, nt, lane);
        c = __builtin_amdgcn_wmma_f32_16x16x32_f16(false, a, false, b, (short)0, c, false, false);
      }
      const int   n  = nt * 16 + nc;
      const float bv = sA3b[n];
#pragma unroll
      for (int r = 0; r < 8; ++r) {
        const int e = e0 + r + 8 * hf;
        if (e < NREL) {
          const float v = c[r] + bv;
          lossLocal += v * v;                               // sum(E_O^2)
          atomicAdd(&nodeAcc1[hidx[e] * 32 + n], v);        // H @ E_O
        }
      }
    }
    // ---- branch B: W1comb + mlp_ee ----
    waveGemmStore<64, 112, true>(bufX, 64, sW1, sB1, bufH, 128, lane);  // cols 0-111
    for (int idx = lane; idx < 16 * 8; idx += 32)                       // zero cols 112-127
      bufH32[(idx >> 3) * 64 + 56 + (idx & 7)] = 0u;
#pragma unroll
    for (int nt = 0; nt < 2; ++nt) {  // E_R = relu(xW1^T+b1) @ ee^T + eeb (N=32)
      v8f c = {};
#pragma unroll
      for (int ks = 0; ks < 4; ++ks) {
        v16h a = ldsAv(bufH, 128, ks, lane);
        v16h b = ldsBswz(sEE, 2, ks, nt, lane);
        c = __builtin_amdgcn_wmma_f32_16x16x32_f16(false, a, false, b, (short)0, c, false, false);
      }
      const int   n  = nt * 16 + nc;
      const float bv = sEEb[n];
#pragma unroll
      for (int r = 0; r < 8; ++r) {
        const int e = e0 + r + 8 * hf;
        if (e < NREL) atomicAdd(&nodeAcc2[hidx[e] * 32 + n], c[r] + bv);  // h2 = E_R^T @ H^T
      }
    }
  }
  atomicAdd(lossAcc, lossLocal);
}

// ---------------------------------------------------------------------------
// Kernel 3 (per phase): gathered-E_R edge MLP (phi_U_R):
//   g = h2[:,recv] + h2[:,send];  loss += |g|^2
//   Ra = relu( relu([E0 | g] @ U1^T + b) @ U2^T + b2 )   (40 ->100 ->8)
// ---------------------------------------------------------------------------
__global__ __launch_bounds__(TPB) void edge_kernel_B(
    const int* __restrict__ recv, const int* __restrict__ send,
    const float* __restrict__ e0feat,   // [NREL][8]
    const float* __restrict__ h2node,   // [NNODES][32]
    const float* U1w, const float* U1b, // 100x40
    const float* U2w, const float* U2b, // 8x100
    float* __restrict__ outRa,          // [NREL][8]
    float* __restrict__ lossAcc) {
  __shared__ __align__(32) _Float16 sU1[64 * 112], sU2[128 * 16];
  __shared__ float sU1b[112], sU2b[16];
  __shared__ __align__(32) _Float16 sXb[WPB][16 * 64];
  __shared__ __align__(32) _Float16 sHb[WPB][16 * 128];

  const int tid = threadIdx.x;
  stageWswz(sU1, U1w, 100, 40, 64, 112, tid);
  stageWswz(sU2, U2w, 8, 100, 128, 16, tid);
  stageB(sU1b, U1b, 100, 112, tid);
  stageB(sU2b, U2b, 8, 16, tid);
  __syncthreads();

  const int lane = tid & 31;
  const int wv   = tid >> 5;
  const int gw   = blockIdx.x * WPB + wv;
  const int nw   = gridDim.x * WPB;
  const int hf   = (lane >> 4) & 1;
  const int nc   = lane & 15;
  _Float16* bufX = sXb[wv];
  _Float16* bufH = sHb[wv];
  unsigned int* bufX32 = (unsigned int*)bufX;
  unsigned int* bufH32 = (unsigned int*)bufH;
  float lossLocal = 0.f;

  for (int tile = gw; tile < NTILES; tile += nw) {
    const int e0 = tile * 16;
    for (int idx = lane; idx < 16 * 32; idx += 32) {
      const int row = idx >> 5, c0 = (idx & 31) << 1;
      const int e = e0 + row;
      float v0 = 0.f, v1 = 0.f;
      if (e < NREL) {
        if (c0 < 8) {
          const float* p = &e0feat[e * 8 + c0];
          v0 = p[0]; v1 = p[1];
        } else if (c0 < 40) {
          const int c = c0 - 8;
          const float* pr = &h2node[recv[e] * 32 + c];
          const float* ps = &h2node[send[e] * 32 + c];
          v0 = pr[0] + ps[0];
          v1 = pr[1] + ps[1];
          lossLocal += v0 * v0 + v1 * v1;          // sum(E_R_gathered^2)
        }
      }
      union { _Float16 h[2]; unsigned int u; } pk;
      pk.h[0] = (_Float16)v0; pk.h[1] = (_Float16)v1;
      bufX32[idx] = pk.u;
    }
    waveGemmStore<64, 112, true>(bufX, 64, sU1, sU1b, bufH, 128, lane);
    for (int idx = lane; idx < 16 * 8; idx += 32)
      bufH32[(idx >> 3) * 64 + 56 + (idx & 7)] = 0u;
    v8f c = {};
#pragma unroll
    for (int ks = 0; ks < 4; ++ks) {
      v16h a = ldsAv(bufH, 128, ks, lane);
      v16h b = ldsBswz(sU2, 1, ks, 0, lane);
      c = __builtin_amdgcn_wmma_f32_16x16x32_f16(false, a, false, b, (short)0, c, false, false);
    }
    const int n = nc;
#pragma unroll
    for (int r = 0; r < 8; ++r) {
      const int e = e0 + r + 8 * hf;
      if (e < NREL && n < 8) outRa[e * 8 + n] = fmaxf(c[r] + sU2b[n], 0.f);
    }
  }
  atomicAdd(lossAcc, lossLocal);
}

// ---------------------------------------------------------------------------
// Kernel 4 (per phase): node MLP + softmax. Tiny (300 rows, ~1 MFLOP).
// ---------------------------------------------------------------------------
__global__ void node_kernel(const float* __restrict__ F0,
                            const float* __restrict__ acc,
                            const float* __restrict__ X0,
                            const float* L1w, const float* L1b,
                            const float* L2w, const float* L2b,
                            float* outA, float* outSm, int reluOut) {
  const int n = blockIdx.x * blockDim.x + threadIdx.x;
  if (n >= NNODES) return;
  float in[51];
  for (int i = 0; i < 16; ++i) in[i] = F0[n * 16 + i];
  for (int i = 0; i < 32; ++i) in[16 + i] = acc[n * 32 + i];
  for (int i = 0; i < 3; ++i)  in[48 + i] = X0[n * 3 + i];
  float h[50];
  for (int j = 0; j < 50; ++j) {
    float s = L1b[j];
    for (int k = 0; k < 51; ++k) s += L1w[j * 51 + k] * in[k];
    h[j] = fmaxf(s, 0.f);
  }
  float o[16], mx = -1e30f;
  for (int j = 0; j < 16; ++j) {
    float s = L2b[j];
    for (int k = 0; k < 50; ++k) s += L2w[j * 50 + k] * h[k];
    if (reluOut) s = fmaxf(s, 0.f);
    o[j] = s;
    mx = fmaxf(mx, s);
    if (outA) outA[n * 16 + j] = s;
  }
  float ex[16], se = 0.f;
  for (int j = 0; j < 16; ++j) { ex[j] = expf(o[j] - mx); se += ex[j]; }
  const float inv = 1.f / se;
  for (int j = 0; j < 16; ++j) outSm[n * 16 + j] = ex[j] * inv;
}

// ---- helpers --------------------------------------------------------------
__global__ void zero_acc_kernel(float* acc1, float* acc2, float* lossAcc, int zeroLoss) {
  const int i = blockIdx.x * blockDim.x + threadIdx.x;
  if (i < NNODES * 32) { acc1[i] = 0.f; acc2[i] = 0.f; }
  if (i == 0 && zeroLoss) lossAcc[0] = 0.f;
}
__global__ void finalize_loss_kernel(const float* lossAcc, float* out) {
  if (blockIdx.x == 0 && threadIdx.x == 0) out[0] = 0.0005f * lossAcc[0];  // 0.001 * 0.5
}

// ---------------------------------------------------------------------------
extern "C" void kernel_launch(void* const* d_in, const int* in_sizes, int n_in,
                              void* d_out, int out_size, void* d_ws, size_t ws_size,
                              hipStream_t stream) {
  (void)in_sizes; (void)n_in; (void)out_size; (void)ws_size;
  // Input order: setup_inputs() dict; params dict flattened with sorted keys,
  // each _lin tuple as (w, b), lists in order.
  const float* R  = (const float*)d_in[0];
  const float* S  = (const float*)d_in[1];
  const float* H  = (const float*)d_in[2];
  const float* E0 = (const float*)d_in[3];
  const float* F0 = (const float*)d_in[4];
  const float* X0 = (const float*)d_in[5];
  /* d_in[6] = X_0 unused */
  const float* b1   = (const float*)d_in[7];
  const float* b2   = (const float*)d_in[8];
  const float* ee1w = (const float*)d_in[9];
  const float* ee1b = (const float*)d_in[10];
  const float* ee3w = (const float*)d_in[11];
  const float* ee3b = (const float*)d_in[12];
  const float* pE1w0 = (const float*)d_in[13]; const float* pE1b0 = (const float*)d_in[14];
  const float* pE1w1 = (const float*)d_in[15]; const float* pE1b1 = (const float*)d_in[16];
  const float* pE1w2 = (const float*)d_in[17]; const float* pE1b2 = (const float*)d_in[18];
  const float* pE2w0 = (const float*)d_in[19]; const float* pE2b0 = (const float*)d_in[20];
  const float* pE2w1 = (const float*)d_in[21]; const float* pE2b1 = (const float*)d_in[22];
  const float* pE2w2 = (const float*)d_in[23]; const float* pE2b2 = (const float*)d_in[24];
  const float* pUO1w0 = (const float*)d_in[25]; const float* pUO1b0 = (const float*)d_in[26];
  const float* pUO1w1 = (const float*)d_in[27]; const float* pUO1b1 = (const float*)d_in[28];
  const float* pUO2w0 = (const float*)d_in[29]; const float* pUO2b0 = (const float*)d_in[30];
  const float* pUO2w1 = (const float*)d_in[31]; const float* pUO2b1 = (const float*)d_in[32];
  const float* pUR1w0 = (const float*)d_in[33]; const float* pUR1b0 = (const float*)d_in[34];
  const float* pUR1w1 = (const float*)d_in[35]; const float* pUR1b1 = (const float*)d_in[36];
  const float* pUR2w0 = (const float*)d_in[37]; const float* pUR2b0 = (const float*)d_in[38];
  const float* pUR2w1 = (const float*)d_in[39]; const float* pUR2b1 = (const float*)d_in[40];
  const float* w1_1 = (const float*)d_in[41];
  const float* w1_2 = (const float*)d_in[42];
  const float* w2_1 = (const float*)d_in[43];
  const float* w2_2 = (const float*)d_in[44];

  float* out   = (float*)d_out;
  float* O3    = out;                       // [300,16]
  float* O3sm  = out + 4800;                // [300,16]
  float* Ra3   = out + 9600;                // [NREL,8]
  float* lossO = out + 9600 + NREL * 8;     // scalar

  char* ws = (char*)d_ws;
  int*   recv = (int*)ws;    ws += (size_t)NREL * sizeof(int);
  int*   send = (int*)ws;    ws += (size_t)NREL * sizeof(int);
  int*   hidx = (int*)ws;    ws += (size_t)NREL * sizeof(int);
  float* acc1 = (float*)ws;  ws += (size_t)NNODES * 32 * sizeof(float);
  float* acc2 = (float*)ws;  ws += (size_t)NNODES * 32 * sizeof(float);
  float* O2   = (float*)ws;  ws += (size_t)NNODES * 16 * sizeof(float);
  float* Ra2  = (float*)ws;  ws += (size_t)NREL * 8 * sizeof(float);
  float* lacc = (float*)ws;

  extract_idx_kernel<<<2048, 256, 0, stream>>>(R, S, H, recv, send, hidx);

  const int zb = (NNODES * 32 + 255) / 256;
  const int eb = 320;  // persistent: 320 blocks x 4 waves over 5607 tiles

  // ---- phase 1 ----
  zero_acc_kernel<<<zb, 256, 0, stream>>>(acc1, acc2, lacc, 1);
  edge_kernel_A<<<eb, TPB, 0, stream>>>(recv, send, hidx, F0, E0,
      pE1w0, pE1b0, pE1w1, pE1b1, pE1w2, pE1b2,
      w1_1, w1_2, b1, ee1w, ee1b, acc1, acc2, lacc);
  node_kernel<<<(NNODES + 63) / 64, 64, 0, stream>>>(F0, acc1, X0,
      pUO1w0, pUO1b0, pUO1w1, pUO1b1, nullptr, O2, 1);       // O_2 = softmax(relu(...))
  edge_kernel_B<<<eb, TPB, 0, stream>>>(recv, send, E0, acc2,
      pUR1w0, pUR1b0, pUR1w1, pUR1b1, Ra2, lacc);            // Ra_2

  // ---- phase 2 ----
  zero_acc_kernel<<<zb, 256, 0, stream>>>(acc1, acc2, lacc, 0);
  edge_kernel_A<<<eb, TPB, 0, stream>>>(recv, send, hidx, O2, Ra2,
      pE2w0, pE2b0, pE2w1, pE2b1, pE2w2, pE2b2,
      w2_1, w2_2, b2, ee3w, ee3b, acc1, acc2, lacc);
  node_kernel<<<(NNODES + 63) / 64, 64, 0, stream>>>(F0, acc1, X0,
      pUO2w0, pUO2b0, pUO2w1, pUO2b1, O3, O3sm, 0);          // O_3, softmax(O_3)
  edge_kernel_B<<<eb, TPB, 0, stream>>>(recv, send, E0, acc2,
      pUR2w0, pUR2b0, pUR2w1, pUR2b1, Ra3, lacc);            // Ra_3

  finalize_loss_kernel<<<1, 32, 0, stream>>>(lacc, lossO);
}
